// BertSelfAttention_78426102825012
// MI455X (gfx1250) — matrix operand order repaired and checked by reading, not compiled, not run on hardware
//
#include <hip/hip_runtime.h>

#define LOG2E 1.44269504088896340736f

typedef __attribute__((ext_vector_type(16))) __bf16 v16bf;
typedef __attribute__((ext_vector_type(8)))  float v8f;

union BV { v16bf v; unsigned u[8]; };

static constexpr int Bb = 4, Ss = 2048, Dd = 1024, Hh = 16, HDd = 64;

__device__ __forceinline__ unsigned short f32_bf16(float f) {
  unsigned x = __float_as_uint(f);
  x += 0x7FFFu + ((x >> 16) & 1u);          // round-to-nearest-even
  return (unsigned short)(x >> 16);
}

// A-matrix 16x32 bf16 tile in this code's contiguous k order (B uses the same): lane m=L%16, h=L/16 ; VGPR v holds K = 16h+2v+{0,1}
__device__ __forceinline__ void load_a_tile(BV& a, const unsigned short* __restrict__ arow,
                                            int k0, int h) {
  #pragma unroll
  for (int v = 0; v < 8; ++v) {
    int kc = k0 + h * 16 + v * 2;
    a.u[v] = *(const unsigned*)(arow + kc);
  }
}

// B-matrix 32x16 bf16 tile in this code's contiguous k order (same as A): lane n=L%16, h=L/16 ; VGPR v holds K = 16h+2v+{0,1}
// wrow must already point at element (K=16h, N=n) with contraction contiguous.
__device__ __forceinline__ void load_b_tile(BV& b, const unsigned short* __restrict__ wrow) {
  #pragma unroll
  for (int v = 0; v < 8; ++v)
    b.u[v] = *(const unsigned*)(wrow + 2 * v);
}

// ---------------------------------------------------------------- cvt f32->bf16
__global__ void cvt_f32_bf16(const float* __restrict__ src,
                             unsigned short* __restrict__ dst, int n4) {
  int i = blockIdx.x * blockDim.x + threadIdx.x;
  if (i < n4) {
    float4 f = ((const float4*)src)[i];
    uint2 o;
    o.x = (unsigned)f32_bf16(f.x) | ((unsigned)f32_bf16(f.y) << 16);
    o.y = (unsigned)f32_bf16(f.z) | ((unsigned)f32_bf16(f.w) << 16);
    ((uint2*)dst)[i] = o;
  }
}

// ---------------------------------------------------------------- QKV GEMM
// Y = X * W^T + b ; one wave computes a 32(M) x 64(N) tile with double-buffered k-loop.
// mode (blockIdx.z): 0=Q (scaled, [bh][s][d]), 1=K ([bh][s][d]), 2=V transposed ([bh][d][s])
__global__ void __launch_bounds__(128) qkv_gemm(
    const unsigned short* __restrict__ Xb,
    const unsigned short* __restrict__ Wqb, const unsigned short* __restrict__ Wkb,
    const unsigned short* __restrict__ Wvb,
    const float* __restrict__ bq, const float* __restrict__ bk, const float* __restrict__ bv,
    unsigned short* __restrict__ Qb, unsigned short* __restrict__ Kb,
    unsigned short* __restrict__ Vtb) {
  const int mode = blockIdx.z;
  const unsigned short* W = (mode == 0) ? Wqb : ((mode == 1) ? Wkb : Wvb);
  const float* bias = (mode == 0) ? bq : ((mode == 1) ? bk : bv);
  const float scale = (mode == 0) ? 0.125f * LOG2E : 1.0f;

  const int tid = threadIdx.x;
  const int wave = tid >> 5;
  const int lane = tid & 31;
  const int h  = lane >> 4;
  const int ln = lane & 15;

  const int m0 = blockIdx.y * 32;
  const int n0 = blockIdx.x * 256 + wave * 64;

  v8f acc[2][4];
  #pragma unroll
  for (int mi = 0; mi < 2; ++mi)
    #pragma unroll
    for (int t = 0; t < 4; ++t) acc[mi][t] = {};

  const unsigned short* arow0 = Xb + (size_t)(m0 + ln) * Dd;
  const unsigned short* arow1 = arow0 + (size_t)16 * Dd;
  const unsigned short* wrow[4];
  #pragma unroll
  for (int t = 0; t < 4; ++t)
    wrow[t] = W + (size_t)(n0 + t * 16 + ln) * Dd + h * 16;

  BV aP[2], bP[4], aQ[2], bQv[4];

  // prologue: k = 0 into buffer P
  load_a_tile(aP[0], arow0, 0, h);
  load_a_tile(aP[1], arow1, 0, h);
  #pragma unroll
  for (int t = 0; t < 4; ++t) load_b_tile(bP[t], wrow[t]);

  for (int k0 = 0; k0 < Dd; k0 += 64) {
    // issue loads for k0+32 into buffer Q before consuming P
    load_a_tile(aQ[0], arow0, k0 + 32, h);
    load_a_tile(aQ[1], arow1, k0 + 32, h);
    #pragma unroll
    for (int t = 0; t < 4; ++t) load_b_tile(bQv[t], wrow[t] + k0 + 32);

    #pragma unroll
    for (int mi = 0; mi < 2; ++mi)
      #pragma unroll
      for (int t = 0; t < 4; ++t)
        acc[mi][t] = __builtin_amdgcn_wmma_f32_16x16x32_bf16(
            false, aP[mi].v, false, bP[t].v, (short)0, acc[mi][t], false, false);

    // issue loads for k0+64 into buffer P before consuming Q
    if (k0 + 64 < Dd) {
      load_a_tile(aP[0], arow0, k0 + 64, h);
      load_a_tile(aP[1], arow1, k0 + 64, h);
      #pragma unroll
      for (int t = 0; t < 4; ++t) load_b_tile(bP[t], wrow[t] + k0 + 64);
    }

    #pragma unroll
    for (int mi = 0; mi < 2; ++mi)
      #pragma unroll
      for (int t = 0; t < 4; ++t)
        acc[mi][t] = __builtin_amdgcn_wmma_f32_16x16x32_bf16(
            false, aQ[mi].v, false, bQv[t].v, (short)0, acc[mi][t], false, false);
  }

  const int hh = n0 / HDd;   // head index (n0 is a multiple of 64)
  #pragma unroll
  for (int mi = 0; mi < 2; ++mi)
    #pragma unroll
    for (int t = 0; t < 4; ++t) {
      float bval = bias[n0 + t * 16 + ln];
      #pragma unroll
      for (int r = 0; r < 8; ++r) {
        int m  = m0 + mi * 16 + r + 8 * h;       // C layout row
        int b_ = m >> 11;                        // / S
        int s_ = m & (Ss - 1);
        int d  = t * 16 + ln;
        unsigned short val = f32_bf16((acc[mi][t][r] + bval) * scale);
        int bh = b_ * Hh + hh;
        if (mode == 2)
          Vtb[((size_t)bh * HDd + d) * Ss + s_] = val;           // V^T
        else if (mode == 0)
          Qb[((size_t)bh * Ss + s_) * HDd + d] = val;
        else
          Kb[((size_t)bh * Ss + s_) * HDd + d] = val;
      }
    }
}

// ---------------------------------------------------------------- attention
// One wave = 16 query rows; flash-style online softmax over 32-key chunks.
__global__ void __launch_bounds__(256) attn_kernel(
    const unsigned short* __restrict__ Qb,
    const unsigned short* __restrict__ Kb,
    const unsigned short* __restrict__ Vtb,
    const float* __restrict__ mask,
    float* __restrict__ out) {
  __shared__ unsigned short Plds[8 * 16 * 32];   // per-wave 16x32 staging

  const int tid = threadIdx.x;
  const int wave = tid >> 5;
  const int lane = tid & 31;
  const int h  = lane >> 4;
  const int ln = lane & 15;

  const int bh = blockIdx.y;
  const int b_ = bh >> 4;          // / H
  const int hh = bh & (Hh - 1);
  const int q0 = blockIdx.x * 128 + wave * 16;

  unsigned short* pbase = Plds + wave * 512;

  // Q A-tiles: 16x64 as two 16x32 (already scaled by 0.125*log2e)
  BV qa[2];
  const unsigned short* qrow = Qb + ((size_t)bh * Ss + q0 + ln) * HDd;
  load_a_tile(qa[0], qrow, 0, h);
  load_a_tile(qa[1], qrow, 32, h);

  float mrow[8], lrow[8];
  v8f acc[4];
  #pragma unroll
  for (int r = 0; r < 8; ++r) { mrow[r] = -3.0e38f; lrow[r] = 0.0f; }
  #pragma unroll
  for (int t = 0; t < 4; ++t) acc[t] = {};

  const unsigned short* kbase = Kb  + (size_t)bh * Ss * HDd;
  const unsigned short* vbase = Vtb + (size_t)bh * HDd * Ss;

  for (int k0 = 0; k0 < Ss; k0 += 32) {
    // prefetch next chunk (K rows are contiguous 4KB; V^T rows strided by S)
    if (k0 + 32 < Ss) {
      __builtin_prefetch(kbase + ((size_t)(k0 + 32) + lane) * HDd, 0, 1);
      __builtin_prefetch(vbase + (size_t)lane * Ss        + k0 + 32, 0, 1);
      __builtin_prefetch(vbase + (size_t)(lane + 32) * Ss + k0 + 32, 0, 1);
    }

    // additive mask (log2 domain)
    float mv0 = mask[(size_t)b_ * Ss + k0 + ln] * LOG2E;
    float mv1 = mask[(size_t)b_ * Ss + k0 + 16 + ln] * LOG2E;

    // issue K B-tile loads (keys k0..+15 and k0+16..+31, head-dim halves)
    BV bk[2][2];
    #pragma unroll
    for (int kt = 0; kt < 2; ++kt)
      #pragma unroll
      for (int dc = 0; dc < 2; ++dc)
        load_b_tile(bk[kt][dc],
                    kbase + ((size_t)k0 + kt * 16 + ln) * HDd + dc * 32 + h * 16);

    // issue V B-tile loads EARLY: independent of scores, latency hidden by softmax
    BV vtile[4];
    #pragma unroll
    for (int t = 0; t < 4; ++t)
      load_b_tile(vtile[t], vbase + ((size_t)t * 16 + ln) * Ss + k0 + h * 16);

    // score WMMAs
    v8f ct0 = {}, ct1 = {};
    #pragma unroll
    for (int dc = 0; dc < 2; ++dc) {
      ct0 = __builtin_amdgcn_wmma_f32_16x16x32_bf16(false, qa[dc].v, false, bk[0][dc].v,
                                                    (short)0, ct0, false, false);
      ct1 = __builtin_amdgcn_wmma_f32_16x16x32_bf16(false, qa[dc].v, false, bk[1][dc].v,
                                                    (short)0, ct1, false, false);
    }

    // online softmax per row (row r+8h lives across the 16 lanes of this half)
    #pragma unroll
    for (int r = 0; r < 8; ++r) {
      float s0 = ct0[r] + mv0;
      float s1 = ct1[r] + mv1;
      float mx = fmaxf(s0, s1);
      #pragma unroll
      for (int off = 1; off < 16; off <<= 1)
        mx = fmaxf(mx, __shfl_xor(mx, off, 32));
      float mnew  = fmaxf(mrow[r], mx);
      float alpha = exp2f(mrow[r] - mnew);
      float p0 = exp2f(s0 - mnew);
      float p1 = exp2f(s1 - mnew);
      float ps = p0 + p1;
      #pragma unroll
      for (int off = 1; off < 16; off <<= 1)
        ps += __shfl_xor(ps, off, 32);
      lrow[r] = lrow[r] * alpha + ps;
      mrow[r] = mnew;
      #pragma unroll
      for (int t = 0; t < 4; ++t) acc[t][r] *= alpha;
      // stage P tile (C layout -> LDS row-major 16x32)
      pbase[(r + 8 * h) * 32 + ln]      = f32_bf16(p0);
      pbase[(r + 8 * h) * 32 + 16 + ln] = f32_bf16(p1);
    }

    // re-read P in A layout (in-wave DS ordering; no barrier needed)
    BV pa;
    load_a_tile(pa, pbase + ln * 32, 0, h);

    // PV WMMAs (V tiles already in registers)
    #pragma unroll
    for (int t = 0; t < 4; ++t)
      acc[t] = __builtin_amdgcn_wmma_f32_16x16x32_bf16(false, pa.v, false, vtile[t].v,
                                                       (short)0, acc[t], false, false);
  }

  // normalize + write ctx back in [B,S,D] f32
  #pragma unroll
  for (int r = 0; r < 8; ++r) {
    float inv = 1.0f / lrow[r];
    int s_ = q0 + r + 8 * h;
    float* orow = out + ((size_t)b_ * Ss + s_) * Dd + hh * HDd;
    #pragma unroll
    for (int t = 0; t < 4; ++t)
      orow[t * 16 + ln] = acc[t][r] * inv;
  }
}

// ---------------------------------------------------------------- launch
extern "C" void kernel_launch(void* const* d_in, const int* in_sizes, int n_in,
                              void* d_out, int out_size, void* d_ws, size_t ws_size,
                              hipStream_t stream) {
  (void)in_sizes; (void)n_in; (void)out_size; (void)ws_size;
  const float* hidden = (const float*)d_in[0];
  const float* mask   = (const float*)d_in[1];
  const float* Wq = (const float*)d_in[2];
  const float* bq = (const float*)d_in[3];
  const float* Wk = (const float*)d_in[4];
  const float* bk = (const float*)d_in[5];
  const float* Wv = (const float*)d_in[6];
  const float* bv = (const float*)d_in[7];
  float* out = (float*)d_out;

  char* ws = (char*)d_ws;
  unsigned short* Xb  = (unsigned short*)(ws + 0);          // 8192x1024 bf16 = 16 MiB
  unsigned short* Wqb = (unsigned short*)(ws + 16777216);   // 1024x1024 bf16 =  2 MiB
  unsigned short* Wkb = (unsigned short*)(ws + 18874368);
  unsigned short* Wvb = (unsigned short*)(ws + 20971520);
  unsigned short* Qbf = (unsigned short*)(ws + 23068672);   // [bh][s][d] bf16 = 16 MiB
  unsigned short* Kbf = (unsigned short*)(ws + 39845888);   // [bh][s][d]
  unsigned short* Vtb = (unsigned short*)(ws + 56623104);   // [bh][d][s] (transposed)

  const int nX4 = (Bb * Ss * Dd) / 4;       // 2,097,152
  const int nW4 = (Dd * Dd) / 4;            //   262,144
  cvt_f32_bf16<<<(nX4 + 255) / 256, 256, 0, stream>>>(hidden, Xb, nX4);
  cvt_f32_bf16<<<(nW4 + 255) / 256, 256, 0, stream>>>(Wq, Wqb, nW4);
  cvt_f32_bf16<<<(nW4 + 255) / 256, 256, 0, stream>>>(Wk, Wkb, nW4);
  cvt_f32_bf16<<<(nW4 + 255) / 256, 256, 0, stream>>>(Wv, Wvb, nW4);

  // M-tiles = 8192/32 = 256, N covered 256 per block, z selects Q/K/V
  qkv_gemm<<<dim3(Dd / 256, (Bb * Ss) / 32, 3), 128, 0, stream>>>(
      Xb, Wqb, Wkb, Wvb, bq, bk, bv, Qbf, Kbf, Vtb);

  // 128 query rows per block (8 waves), 64 (b,h) pairs
  attn_kernel<<<dim3(Ss / 128, Bb * Hh), 256, 0, stream>>>(Qbf, Kbf, Vtb, mask, out);
}
